// LSTMPolicyMultiDiscrete__8985071583424
// MI455X (gfx1250) — matrix operand order repaired and checked by reading, not compiled, not run on hardware
//
#include <hip/hip_runtime.h>
#include <hip/hip_bf16.h>
#include <math.h>

// ---------------------------------------------------------------------------
// MI455X / gfx1250, wave32. Matrix core path: v_wmma_f32_16x16x32_bf16.
// LSTM GEMM: LDS-staged weight tiles (bank-conflict-free padded layout),
// register double-buffered staging, and 2x M-tile B-fragment reuse so each
// LDS fragment feeds two WMMAs.
// ---------------------------------------------------------------------------

typedef __attribute__((ext_vector_type(16))) __bf16 bf16x16;
typedef __attribute__((ext_vector_type(8)))  __bf16 bf16x8;
typedef __attribute__((ext_vector_type(8)))  float  f32x8;

#define PI_F 3.14159265358979323846f

// Sizes (match reference)
#define BB   2048
#define SS   5
#define INF_ 256
#define HH   512
#define NTAU 8
#define NROW (BB*NTAU)      // 16384 LSTM batch rows
#define MCOS (BB*SS*NTAU)   // 81920 cos-feature rows
#define MHEAD (BB*SS)       // 10240 head rows
#define OUTD 36

// LSTM LDS staging: K-chunk of 64, 128 weight rows (4 gates x 32 n),
// rows padded to 72 bf16 (144 B = 36 dwords) for bank spread.
#define KC     64
#define LDSROW 72
#define WROWS  128
#define MBLK   256           // block M tile (2 x 128, two m-tiles per wave)

// ---------------------------------------------------------------------------
// WMMA fragment loaders (bf16, contiguous k = 16*half + s in both A and B, wave32)
// ---------------------------------------------------------------------------

// A-matrix 16x32 bf16 (MxK): lane = m + 16*half; lane holds K = k0+half*16+0..15
// contiguous (matches B fragments). Two 16B loads.
__device__ __forceinline__ bf16x16 load_a_frag(const __bf16* base, int m0,
                                               int lda, int k0, int lane) {
    int mm   = lane & 15;
    int half = lane >> 4;
    const __bf16* p = base + (size_t)(m0 + mm) * lda + k0 + half * 16;
    bf16x8 lo = *(const bf16x8*)(p);
    bf16x8 hi = *(const bf16x8*)(p + 8);
    return __builtin_shufflevector(lo, hi, 0,1,2,3,4,5,6,7,8,9,10,11,12,13,14,15);
}

// B-matrix 32x16 bf16 (KxN), sourced from row-major weight W[n][k] (i.e. X@W^T):
// lane = n + 16*half; lane holds K = k0 + half*16 + 0..15 contiguous (32B).
__device__ __forceinline__ bf16x16 load_b_frag(const __bf16* w, int n0,
                                               int ldb, int k0, int lane) {
    int nn   = lane & 15;
    int half = lane >> 4;
    const __bf16* p = w + (size_t)(n0 + nn) * ldb + k0 + half * 16;
    bf16x8 lo = *(const bf16x8*)(p);
    bf16x8 hi = *(const bf16x8*)(p + 8);
    return __builtin_shufflevector(lo, hi, 0,1,2,3,4,5,6,7,8,9,10,11,12,13,14,15);
}

// B fragment from LDS-staged tile (row-major rows of LDSROW bf16).
__device__ __forceinline__ bf16x16 load_b_lds(const __bf16* sm, int row_base,
                                              int k_local, int lane) {
    int nn   = lane & 15;
    int half = lane >> 4;
    const __bf16* p = sm + (row_base + nn) * LDSROW + k_local + half * 16;
    bf16x8 lo = *(const bf16x8*)(p);
    bf16x8 hi = *(const bf16x8*)(p + 8);
    return __builtin_shufflevector(lo, hi, 0,1,2,3,4,5,6,7,8,9,10,11,12,13,14,15);
}

__device__ __forceinline__ f32x8 wmma_bf16(bf16x16 a, bf16x16 b, f32x8 c) {
    return __builtin_amdgcn_wmma_f32_16x16x32_bf16(false, a, false, b,
                                                   (short)0, c, false, false);
}

__device__ __forceinline__ float sigm(float x) {
    return 1.0f / (1.0f + __expf(-x));
}

// ---------------------------------------------------------------------------
// Small helper kernels
// ---------------------------------------------------------------------------

__global__ void k_cvt_bf16(const float* __restrict__ src,
                           __bf16* __restrict__ dst, int n) {
    int i = blockIdx.x * blockDim.x + threadIdx.x;
    if (i < n) dst[i] = (__bf16)src[i];
}

__global__ void k_addv(const float* __restrict__ a, const float* __restrict__ b,
                       float* __restrict__ o, int n) {
    int i = blockIdx.x * blockDim.x + threadIdx.x;
    if (i < n) o[i] = a[i] + b[i];
}

// cos features: cosf[m][c] = cos(tau[m] * pi * c), m < 81920, c < 64, bf16 out
__global__ void k_cosfeat(const float* __restrict__ taus,
                          __bf16* __restrict__ out, int n) {
    int i = blockIdx.x * blockDim.x + threadIdx.x;
    if (i >= n) return;
    int m = i >> 6;
    int c = i & 63;
    out[i] = (__bf16)__cosf(taus[m] * (PI_F * (float)c));
}

// ---------------------------------------------------------------------------
// WMMA GEMM kernels (head / cos): block 256 thr = 8 waves, wave tile 16x16.
// ---------------------------------------------------------------------------

// head: h = relu(X @ head_w^T + b), M=10240, K=256, N=512, bf16 out
__global__ void k_head(const __bf16* __restrict__ xa,
                       const __bf16* __restrict__ wb,
                       const float* __restrict__ bias,
                       __bf16* __restrict__ out) {
    int lane = threadIdx.x & 31;
    int w    = threadIdx.x >> 5;
    int m0 = blockIdx.x * 64 + (w & 3) * 16;
    int n0 = blockIdx.y * 32 + (w >> 2) * 16;
    f32x8 acc = {};
    for (int k0 = 0; k0 < INF_; k0 += 32) {
        bf16x16 a = load_a_frag(xa, m0, INF_, k0, lane);
        bf16x16 b = load_b_frag(wb, n0, INF_, k0, lane);
        acc = wmma_bf16(a, b, acc);
    }
    int nn = n0 + (lane & 15);
    int half = lane >> 4;
    float bv = bias[nn];
#pragma unroll
    for (int r = 0; r < 8; ++r) {
        int mm = m0 + half * 8 + r;
        float v = acc[r] + bv;
        v = v > 0.0f ? v : 0.0f;
        out[(size_t)mm * HH + nn] = (__bf16)v;
    }
}

// cos_x GEMM fused with broadcast-multiply by head features:
// P[m][n] = relu(cosf[m] @ cos_w^T + b)[n] * hfeat[m/8][n]; M=81920, K=64
__global__ void k_cosx(const __bf16* __restrict__ ca,
                       const __bf16* __restrict__ wb,
                       const float* __restrict__ bias,
                       const __bf16* __restrict__ hfeat,
                       __bf16* __restrict__ P) {
    int lane = threadIdx.x & 31;
    int w    = threadIdx.x >> 5;
    int m0 = blockIdx.x * 64 + (w & 3) * 16;
    int n0 = blockIdx.y * 32 + (w >> 2) * 16;
    f32x8 acc = {};
    for (int k0 = 0; k0 < 64; k0 += 32) {
        bf16x16 a = load_a_frag(ca, m0, 64, k0, lane);
        bf16x16 b = load_b_frag(wb, n0, 64, k0, lane);
        acc = wmma_bf16(a, b, acc);
    }
    int nn = n0 + (lane & 15);
    int half = lane >> 4;
    float bv = bias[nn];
#pragma unroll
    for (int r = 0; r < 8; ++r) {
        int mm = m0 + half * 8 + r;
        float v = acc[r] + bv;
        v = v > 0.0f ? v : 0.0f;
        float hf = (float)hfeat[(size_t)(mm >> 3) * HH + nn];
        P[(size_t)mm * HH + nn] = (__bf16)(v * hf);
    }
}

// ---------------------------------------------------------------------------
// Fused LSTM step with LDS weight staging + double buffering.
// Block = 256 thr / 8 waves. Block tile: M=256 (wave w -> rows w*16 and
// 128+w*16), N=32 in H-space x 4 gates. Each wave: 16 accumulator sets
// (2 m-tiles x 2 n-tiles x 4 gates); each LDS B fragment feeds 2 WMMAs and
// each A fragment feeds 8. Weight K-chunks (16 KB) staged to LDS, prefetched
// into registers one chunk ahead -> B fetched from L2 once per 256 M-rows.
// ---------------------------------------------------------------------------
__global__ void k_lstm_step(const __bf16* __restrict__ xseq,
                            const __bf16* __restrict__ h_in,
                            const __bf16* __restrict__ Wih,
                            const __bf16* __restrict__ Whh,
                            const float* __restrict__ bsum,
                            const int* __restrict__ lengths,
                            float* __restrict__ cbuf,
                            __bf16* __restrict__ h_out,
                            __bf16* __restrict__ outseq,
                            int t) {
    __shared__ __bf16 smw[WROWS * LDSROW];   // 18 KB

    int tid  = threadIdx.x;
    int lane = tid & 31;
    int w    = tid >> 5;
    int m0   = blockIdx.x * MBLK + w * 16;   // second m-tile at m0 + 128
    int n0b  = blockIdx.y * 32;              // block n-base within H

    f32x8 acc[16];                           // [(gate*2 + n_tile)*2 + m_tile]
#pragma unroll
    for (int i = 0; i < 16; ++i) acc[i] = (f32x8){};

    const __bf16* aptrs[2] = { xseq + (size_t)t * HH, h_in };
    const int     ldas[2]  = { SS * HH, HH };
    const __bf16* wptrs[2] = { Wih, Whh };

    // staging indices for this thread: 4 x 16B vectors of the 16 KB tile
    int srow[4], grow[4], skv[4];
#pragma unroll
    for (int i = 0; i < 4; ++i) {
        int idx = tid + i * 256;             // 0..1023
        srow[i] = idx >> 3;                  // 0..127 (gate*32 + local n)
        skv[i]  = idx & 7;                   // 0..7 (16B column within chunk)
        grow[i] = (srow[i] >> 5) * HH + n0b + (srow[i] & 31); // weight row
    }

    // prefetch chunk 0 (phase 0, kc 0) into registers
    bf16x8 pre[4];
#pragma unroll
    for (int i = 0; i < 4; ++i)
        pre[i] = *(const bf16x8*)(wptrs[0] + (size_t)grow[i] * HH + skv[i] * 8);

    // 16 chunks: cc = phase*8 + kc/KC
    for (int cc = 0; cc < 16; ++cc) {
        int phase = cc >> 3;
        int kc    = (cc & 7) * KC;

        __syncthreads();                     // LDS free from previous chunk
#pragma unroll
        for (int i = 0; i < 4; ++i)
            *(bf16x8*)(smw + srow[i] * LDSROW + skv[i] * 8) = pre[i];
        __syncthreads();                     // tile visible

        // prefetch next chunk while computing this one
        if (cc < 15) {
            int nc  = cc + 1;
            int np  = nc >> 3;
            int nkc = (nc & 7) * KC;
            const __bf16* wp = wptrs[np];
#pragma unroll
            for (int i = 0; i < 4; ++i)
                pre[i] = *(const bf16x8*)(wp + (size_t)grow[i] * HH + nkc + skv[i] * 8);
        }

        const __bf16* ap = aptrs[phase];
        const int lda    = ldas[phase];
#pragma unroll
        for (int ks = 0; ks < KC; ks += 32) {
            bf16x16 a0 = load_a_frag(ap, m0,       lda, kc + ks, lane);
            bf16x16 a1 = load_a_frag(ap, m0 + 128, lda, kc + ks, lane);
#pragma unroll
            for (int g = 0; g < 4; ++g) {
#pragma unroll
                for (int nt = 0; nt < 2; ++nt) {
                    bf16x16 b = load_b_lds(smw, g * 32 + nt * 16, ks, lane);
                    int ai = (g * 2 + nt) * 2;
                    acc[ai + 0] = wmma_bf16(a0, b, acc[ai + 0]);
                    acc[ai + 1] = wmma_bf16(a1, b, acc[ai + 1]);
                }
            }
        }
    }

    // ---- epilogue: bias + gate nonlinearities + masked cell update ----
    int half = lane >> 4;
#pragma unroll
    for (int mt = 0; mt < 2; ++mt) {
#pragma unroll
        for (int nt = 0; nt < 2; ++nt) {
            int nn = n0b + nt * 16 + (lane & 15);
            float bi  = bsum[0 * HH + nn];
            float bf_ = bsum[1 * HH + nn];
            float bg  = bsum[2 * HH + nn];
            float bo  = bsum[3 * HH + nn];
#pragma unroll
            for (int r = 0; r < 8; ++r) {
                int mm = m0 + mt * 128 + half * 8 + r;
                bool mtk = t < lengths[mm >> 3];    // lens repeated NTAU times
                float gi = sigm(acc[(0 * 2 + nt) * 2 + mt][r] + bi);
                float gf = sigm(acc[(1 * 2 + nt) * 2 + mt][r] + bf_);
                float gg = tanhf(acc[(2 * 2 + nt) * 2 + mt][r] + bg);
                float go = sigm(acc[(3 * 2 + nt) * 2 + mt][r] + bo);
                size_t idx = (size_t)mm * HH + nn;
                float c_old = cbuf[idx];
                float c_new = gf * c_old + gi * gg;
                float h_new = go * tanhf(c_new);
                __bf16 h_old = h_in[idx];
                cbuf[idx]  = mtk ? c_new : c_old;
                h_out[idx] = mtk ? (__bf16)h_new : h_old;
                outseq[((size_t)mm * SS + t) * HH + nn] =
                    mtk ? (__bf16)h_new : (__bf16)0.0f;
            }
        }
    }
}

// ---------------------------------------------------------------------------
// Tail kernels (small, plain f32)
// ---------------------------------------------------------------------------

// lo[b][s][n] = mean over tau of out1[(b*8+tau)][s][n]
__global__ void k_mean_seq(const __bf16* __restrict__ out1,
                           float* __restrict__ lo, int n) {
    int i = blockIdx.x * blockDim.x + threadIdx.x;
    if (i >= n) return;
    int nn = i & (HH - 1);
    int s  = (i / HH) % SS;
    int b  = i / (HH * SS);
    size_t base = ((size_t)(b * NTAU) * SS + s) * HH + nn;
    float acc = 0.0f;
#pragma unroll
    for (int tau = 0; tau < NTAU; ++tau)
        acc += (float)out1[base + (size_t)tau * SS * HH];
    lo[i] = acc * (1.0f / NTAU);
}

// q0[b][n] = mean over tau of hN[(b*8+tau)][n]
__global__ void k_mean_h(const __bf16* __restrict__ hN,
                         float* __restrict__ q0, int n) {
    int i = blockIdx.x * blockDim.x + threadIdx.x;
    if (i >= n) return;
    int nn = i & (HH - 1);
    int b  = i / HH;
    size_t base = (size_t)(b * NTAU) * HH + nn;
    float acc = 0.0f;
#pragma unroll
    for (int tau = 0; tau < NTAU; ++tau)
        acc += (float)hN[base + (size_t)tau * HH];
    q0[i] = acc * (1.0f / NTAU);
}

// out[m][n] = in[m] . W[n] + bias[n]   (row-major W[N][K], K mult of 4)
__global__ void k_linear(const float* __restrict__ in,
                         const float* __restrict__ W,
                         const float* __restrict__ bias,
                         float* __restrict__ out, int M, int N, int K) {
    int i = blockIdx.x * blockDim.x + threadIdx.x;
    if (i >= M * N) return;
    int m = i / N, n = i - m * N;
    const float4* a4 = (const float4*)(in + (size_t)m * K);
    const float4* w4 = (const float4*)(W + (size_t)n * K);
    float s = 0.0f;
    for (int k = 0; k < K / 4; ++k) {
        float4 x = a4[k], y = w4[k];
        s += x.x * y.x + x.y * y.y + x.z * y.z + x.w * y.w;
    }
    out[i] = s + bias[n];
}

// Per-batch masked softmax attention: scores -> softmax -> context
__global__ void k_attn_ctx(const float* __restrict__ lo,
                           const float* __restrict__ q,
                           const int* __restrict__ lengths,
                           float* __restrict__ ctx) {
    int b = blockIdx.x;
    int tid = threadIdx.x;
    const float* lob = lo + (size_t)b * SS * HH;
    const float* qb  = q + (size_t)b * HH;
    __shared__ float red[256];
    __shared__ float sw[SS];

    float part[SS];
#pragma unroll
    for (int s = 0; s < SS; ++s) {
        float p = 0.0f;
        for (int k = tid; k < HH; k += 256)
            p += lob[s * HH + k] * qb[k];
        part[s] = p;
    }
#pragma unroll
    for (int s = 0; s < SS; ++s) {
        red[tid] = part[s];
        __syncthreads();
        for (int off = 128; off > 0; off >>= 1) {
            if (tid < off) red[tid] += red[tid + off];
            __syncthreads();
        }
        if (tid == 0) sw[s] = red[0];
        __syncthreads();
    }
    if (tid == 0) {
        int len = lengths[b];
        float mx = -1e30f;
        for (int s = 0; s < len; ++s) mx = fmaxf(mx, sw[s]);
        float e[SS], sum = 0.0f;
        for (int s = 0; s < SS; ++s) {
            e[s] = (s < len) ? __expf(sw[s] - mx) : 0.0f;
            sum += e[s];
        }
        float inv = 1.0f / sum;
        for (int s = 0; s < SS; ++s) sw[s] = e[s] * inv;
    }
    __syncthreads();
    for (int nn = tid; nn < HH; nn += 256) {
        float cv = 0.0f;
#pragma unroll
        for (int s = 0; s < SS; ++s) cv += sw[s] * lob[s * HH + nn];
        ctx[(size_t)b * HH + nn] = cv;
    }
}

// ---------------------------------------------------------------------------
// Host launcher
// ---------------------------------------------------------------------------

extern "C" void kernel_launch(void* const* d_in, const int* in_sizes, int n_in,
                              void* d_out, int out_size, void* d_ws, size_t ws_size,
                              hipStream_t stream) {
    (void)in_sizes; (void)n_in; (void)out_size; (void)ws_size;

    const float* x      = (const float*)d_in[0];
    const float* taus   = (const float*)d_in[1];
    const int*   lens   = (const int*)  d_in[2];
    const float* head_w = (const float*)d_in[3];
    const float* head_b = (const float*)d_in[4];
    const float* cos_w  = (const float*)d_in[5];
    const float* cos_b  = (const float*)d_in[6];
    const float* Wih0   = (const float*)d_in[7];
    const float* Whh0   = (const float*)d_in[8];
    const float* bih0   = (const float*)d_in[9];
    const float* bhh0   = (const float*)d_in[10];
    const float* Wih1   = (const float*)d_in[11];
    const float* Whh1   = (const float*)d_in[12];
    const float* bih1   = (const float*)d_in[13];
    const float* bhh1   = (const float*)d_in[14];
    const float* attn_w = (const float*)d_in[15];
    const float* attn_b = (const float*)d_in[16];
    const float* hid_w  = (const float*)d_in[17];
    const float* hid_b  = (const float*)d_in[18];
    const float* out_w  = (const float*)d_in[19];
    const float* out_b  = (const float*)d_in[20];
    float* outp = (float*)d_out;

    // workspace carve-up (256B aligned)
    char* ws = (char*)d_ws;
    size_t off = 0;
    auto carve = [&](size_t bytes) -> char* {
        char* p = ws + off;
        off = (off + bytes + 255) & ~(size_t)255;
        return p;
    };
    __bf16* xbf   = (__bf16*)carve((size_t)MHEAD * INF_ * 2);
    __bf16* hwbf  = (__bf16*)carve((size_t)HH * INF_ * 2);
    __bf16* cwbf  = (__bf16*)carve((size_t)HH * 64 * 2);
    __bf16* w0i   = (__bf16*)carve((size_t)4 * HH * HH * 2);
    __bf16* w0h   = (__bf16*)carve((size_t)4 * HH * HH * 2);
    __bf16* w1i   = (__bf16*)carve((size_t)4 * HH * HH * 2);
    __bf16* w1h   = (__bf16*)carve((size_t)4 * HH * HH * 2);
    float*  bs0   = (float*) carve((size_t)4 * HH * 4);
    float*  bs1   = (float*) carve((size_t)4 * HH * 4);
    __bf16* hfeat = (__bf16*)carve((size_t)MHEAD * HH * 2);
    __bf16* cfeat = (__bf16*)carve((size_t)MCOS * 64 * 2);
    __bf16* P     = (__bf16*)carve((size_t)MCOS * HH * 2);   // xx; reused as out1
    __bf16* out0  = (__bf16*)carve((size_t)MCOS * HH * 2);
    __bf16* h0a   = (__bf16*)carve((size_t)NROW * HH * 2);
    __bf16* h0b   = (__bf16*)carve((size_t)NROW * HH * 2);
    float*  c0    = (float*) carve((size_t)NROW * HH * 4);
    __bf16* h1a   = (__bf16*)carve((size_t)NROW * HH * 2);
    __bf16* h1b   = (__bf16*)carve((size_t)NROW * HH * 2);
    float*  c1    = (float*) carve((size_t)NROW * HH * 4);
    float*  lo    = (float*) carve((size_t)BB * SS * HH * 4);
    float*  q0    = (float*) carve((size_t)BB * HH * 4);
    float*  q     = (float*) carve((size_t)BB * HH * 4);
    float*  ctx   = (float*) carve((size_t)BB * HH * 4);
    float*  ctx2  = (float*) carve((size_t)BB * HH * 4);
    __bf16* out1  = P;  // alias: P consumed by layer0 before layer1 writes

    const int T = 256;
    auto blocks = [](size_t n, int t) { return (unsigned)((n + t - 1) / t); };

    // 1) precision conversions + bias folding
    k_cvt_bf16<<<blocks((size_t)MHEAD * INF_, T), T, 0, stream>>>(x, xbf, MHEAD * INF_);
    k_cvt_bf16<<<blocks((size_t)HH * INF_, T), T, 0, stream>>>(head_w, hwbf, HH * INF_);
    k_cvt_bf16<<<blocks((size_t)HH * 64, T), T, 0, stream>>>(cos_w, cwbf, HH * 64);
    k_cvt_bf16<<<blocks((size_t)4 * HH * HH, T), T, 0, stream>>>(Wih0, w0i, 4 * HH * HH);
    k_cvt_bf16<<<blocks((size_t)4 * HH * HH, T), T, 0, stream>>>(Whh0, w0h, 4 * HH * HH);
    k_cvt_bf16<<<blocks((size_t)4 * HH * HH, T), T, 0, stream>>>(Wih1, w1i, 4 * HH * HH);
    k_cvt_bf16<<<blocks((size_t)4 * HH * HH, T), T, 0, stream>>>(Whh1, w1h, 4 * HH * HH);
    k_addv<<<blocks(4 * HH, T), T, 0, stream>>>(bih0, bhh0, bs0, 4 * HH);
    k_addv<<<blocks(4 * HH, T), T, 0, stream>>>(bih1, bhh1, bs1, 4 * HH);

    // 2) head GEMM (WMMA): hfeat = relu(x @ head_w^T + b)
    k_head<<<dim3(MHEAD / 64, HH / 32), T, 0, stream>>>(xbf, hwbf, head_b, hfeat);

    // 3) cos features + cos GEMM fused with broadcast multiply -> P (= xx)
    k_cosfeat<<<blocks((size_t)MCOS * 64, T), T, 0, stream>>>(taus, cfeat, MCOS * 64);
    k_cosx<<<dim3(MCOS / 64, HH / 32), T, 0, stream>>>(cfeat, cwbf, cos_b, hfeat, P);

    // 4) LSTM layer 0 (5 fused WMMA steps, ping-pong h)
    hipMemsetAsync(h0a, 0, (size_t)NROW * HH * 2, stream);
    hipMemsetAsync(c0, 0, (size_t)NROW * HH * 4, stream);
    {
        __bf16* hin = h0a; __bf16* hout = h0b;
        for (int t = 0; t < SS; ++t) {
            k_lstm_step<<<dim3(NROW / MBLK, HH / 32), T, 0, stream>>>(
                P, hin, w0i, w0h, bs0, lens, c0, hout, out0, t);
            __bf16* tmp = hin; hin = hout; hout = tmp;
        }
    }

    // 5) LSTM layer 1 (reads out0, writes out1 = P alias)
    hipMemsetAsync(h1a, 0, (size_t)NROW * HH * 2, stream);
    hipMemsetAsync(c1, 0, (size_t)NROW * HH * 4, stream);
    __bf16* hN = h1a;
    {
        __bf16* hin = h1a; __bf16* hout = h1b;
        for (int t = 0; t < SS; ++t) {
            k_lstm_step<<<dim3(NROW / MBLK, HH / 32), T, 0, stream>>>(
                out0, hin, w1i, w1h, bs1, lens, c1, hout, out1, t);
            __bf16* tmp = hin; hin = hout; hout = tmp;
        }
        hN = hin;  // last written buffer
    }

    // 6) tau means
    k_mean_seq<<<blocks((size_t)BB * SS * HH, T), T, 0, stream>>>(out1, lo, BB * SS * HH);
    k_mean_h<<<blocks((size_t)BB * HH, T), T, 0, stream>>>(hN, q0, BB * HH);

    // 7) attention query, softmax context, output head
    k_linear<<<blocks((size_t)BB * HH, T), T, 0, stream>>>(q0, attn_w, attn_b, q, BB, HH, HH);
    k_attn_ctx<<<BB, T, 0, stream>>>(lo, q, lens, ctx);
    k_linear<<<blocks((size_t)BB * HH, T), T, 0, stream>>>(ctx, hid_w, hid_b, ctx2, BB, HH, HH);
    k_linear<<<blocks((size_t)BB * OUTD, T), T, 0, stream>>>(ctx2, out_w, out_b, outp, BB, OUTD, HH);
}
